// EncoderEGNCA_30940944401183
// MI455X (gfx1250) — matrix operand order repaired and checked by reading, not compiled, not run on hardware
//
#include <hip/hip_runtime.h>
#include <hip/hip_bf16.h>
#include <math.h>

typedef __attribute__((ext_vector_type(16))) _Float16 v16h;
typedef __attribute__((ext_vector_type(8)))  _Float16 v8h;
typedef __attribute__((ext_vector_type(8)))  float    v8f;

#define N_NODES 50000
#define N_EDGES 1600000
#define ND 16
#define MD 32
#define NODE_EPS 1e-5f
#define WAVES_PER_BLOCK 8

// ---------------------------------------------------------------------------
// helpers
// ---------------------------------------------------------------------------
__device__ __forceinline__ void lds_fence() {
  // wave-internal LDS store->load ordering (cross-lane data through LDS)
  asm volatile("s_wait_dscnt 0" ::: "memory");
}

// Branch-free tanh on the hardware transcendental path:
//   tanh(x) = 1 - 2 / (exp(2x) + 1) = 1 - 2 * rcp(exp2(x * 2*log2(e)) + 1)
// Saturates correctly at +-1 via inf/0 behavior of v_exp_f32 / v_rcp_f32.
__device__ __forceinline__ float tanh_fast(float x) {
  const float e = __builtin_amdgcn_exp2f(x * 2.885390081777927f);
  return 1.0f - 2.0f * __builtin_amdgcn_rcpf(e + 1.0f);
}

// Build a 32x16 f16 B-fragment (K=32) from a row-major [K x ncols] f32 weight.
// Layout: lane L holds column N=(L&15)+16*ntile; lanes 0-15 hold K=0..7,16..23,
// lanes 16-31 hold K=8..15,24..31 (half i <-> K = kb+i+(i>=8?8:0)).
__device__ __forceinline__ v16h load_Bfrag(const float* __restrict__ W,
                                           int ncols, int ntile, int lane) {
  const int n  = (lane & 15) + ntile * 16;
  const int kb = (lane < 16) ? 0 : 8;
  v16h b;
#pragma unroll
  for (int i = 0; i < 16; ++i)
    b[i] = (_Float16)W[(kb + i + (i >= 8 ? 8 : 0)) * ncols + n];
  return b;
}

__device__ __forceinline__ v8f wmma_f16(v16h a, v16h b, v8f c) {
  return __builtin_amdgcn_wmma_f32_16x16x32_f16(false, a, false, b,
                                                (short)0, c, false, false);
}

// ---------------------------------------------------------------------------
// zero scratch
// ---------------------------------------------------------------------------
__global__ void egnca_zero_kernel(float* __restrict__ p, int n) {
  int i = blockIdx.x * blockDim.x + threadIdx.x;
  if (i < n) p[i] = 0.0f;
}

// ---------------------------------------------------------------------------
// edge kernel: one wave processes 16 edges per tile.
// Edge MLP layers as v_wmma_f32_16x16x32_f16 (K = 2*ND = 32 exactly).
// ---------------------------------------------------------------------------
__global__ __launch_bounds__(256) void egnca_edge_kernel(
    const int* __restrict__ row, const int* __restrict__ col,
    const float* __restrict__ coord, const float* __restrict__ h,
    const float* __restrict__ Wm1, const float* __restrict__ bm1,
    const float* __restrict__ Wm2, const float* __restrict__ bm2,
    const float* __restrict__ Wc1, const float* __restrict__ bc1,
    const float* __restrict__ Wc2,
    float* __restrict__ num, float* __restrict__ cnt, float* __restrict__ agg,
    const int addCnt)
{
  __shared__ __align__(16) _Float16 s_act[WAVES_PER_BLOCK][16 * MD];
  __shared__ float s_d2[WAVES_PER_BLOCK][16];
  __shared__ int   s_row[WAVES_PER_BLOCK][16];
  __shared__ float s_cw[WAVES_PER_BLOCK][16];

  const int lane   = threadIdx.x & 31;
  const int wv     = threadIdx.x >> 5;
  const int nsub   = lane & 15;           // tile column / edge sub-index
  const bool hiH   = lane >= 16;
  const int kb     = hiH ? 8 : 0;         // A-frag K chunk base
  const int mofs   = hiH ? 8 : 0;         // D rows owned by this lane

  // loop-invariant weight fragments (kept in VGPRs)
  const v16h B_m1_0 = load_Bfrag(Wm1, MD, 0, lane);
  const v16h B_m1_1 = load_Bfrag(Wm1, MD, 1, lane);
  const v16h B_m2_0 = load_Bfrag(Wm2, MD, 0, lane);
  const v16h B_m2_1 = load_Bfrag(Wm2, MD, 1, lane);
  const v16h B_c1_0 = load_Bfrag(Wc1, MD, 0, lane);
  const v16h B_c1_1 = load_Bfrag(Wc1, MD, 1, lane);

  const int n0 = nsub, n1 = nsub + 16;    // this lane's output columns
  const float bm1_0 = bm1[n0], bm1_1 = bm1[n1];
  const float wl_0  = Wm1[32 * MD + n0];  // d2 row (33rd input) of Wm1
  const float wl_1  = Wm1[32 * MD + n1];
  const float bm2_0 = bm2[n0], bm2_1 = bm2[n1];
  const float bc1_0 = bc1[n0], bc1_1 = bc1[n1];
  const float wc2_0 = Wc2[n0], wc2_1 = Wc2[n1];

  const int ntiles      = (N_EDGES + 15) / 16;
  const int wave_gid    = blockIdx.x * WAVES_PER_BLOCK + wv;
  const int wave_stride = gridDim.x * WAVES_PER_BLOCK;

  for (int tile = wave_gid; tile < ntiles; tile += wave_stride) {
    const int e_base = tile * 16;
    const int nv     = min(16, N_EDGES - e_base);
    const int eid    = e_base + ((nsub < nv) ? nsub : 0);
    const int ri = row[eid];
    const int ci = col[eid];

    // per-edge geometry (both half-waves compute the same edge)
    const float rx = coord[ri * 3 + 0] - coord[ci * 3 + 0];
    const float ry = coord[ri * 3 + 1] - coord[ci * 3 + 1];
    const float rz = coord[ri * 3 + 2] - coord[ci * 3 + 2];
    const float d2 = rx * rx + ry * ry + rz * rz;
    const float rinv = __builtin_amdgcn_rcpf(__builtin_amdgcn_sqrtf(d2) + 1.0f);
    if (!hiH) {
      s_d2[wv][nsub]  = d2;
      s_row[wv][nsub] = ri;
    }

    // ---- A0 = [h_i | h_j] directly from global (16 edges x 32 f16) ----
    v16h a;
    {
      const float4 hr0 = *(const float4*)(h + ri * ND + kb);
      const float4 hr1 = *(const float4*)(h + ri * ND + kb + 4);
      const float4 hc0 = *(const float4*)(h + ci * ND + kb);
      const float4 hc1 = *(const float4*)(h + ci * ND + kb + 4);
      a[0]  = (_Float16)hr0.x; a[1]  = (_Float16)hr0.y;
      a[2]  = (_Float16)hr0.z; a[3]  = (_Float16)hr0.w;
      a[4]  = (_Float16)hr1.x; a[5]  = (_Float16)hr1.y;
      a[6]  = (_Float16)hr1.z; a[7]  = (_Float16)hr1.w;
      a[8]  = (_Float16)hc0.x; a[9]  = (_Float16)hc0.y;
      a[10] = (_Float16)hc0.z; a[11] = (_Float16)hc0.w;
      a[12] = (_Float16)hc1.x; a[13] = (_Float16)hc1.y;
      a[14] = (_Float16)hc1.z; a[15] = (_Float16)hc1.w;
    }
    lds_fence();                          // d2/row visible wave-wide

    // ---- layer 1: C preloaded with bias + d2 * Wm1[32,:]  (folds 33rd input)
    v8f c0, c1;
#pragma unroll
    for (int r = 0; r < 8; ++r) {
      const float d2m = s_d2[wv][mofs + r];
      c0[r] = bm1_0 + d2m * wl_0;
      c1[r] = bm1_1 + d2m * wl_1;
    }
    c0 = wmma_f16(a, B_m1_0, c0);
    c1 = wmma_f16(a, B_m1_1, c1);

    // tanh -> t1 -> LDS (row-major 16x32 halves) for D->A transpose
#pragma unroll
    for (int r = 0; r < 8; ++r) {
      s_act[wv][(mofs + r) * MD + n0] = (_Float16)tanh_fast(c0[r]);
      s_act[wv][(mofs + r) * MD + n1] = (_Float16)tanh_fast(c1[r]);
    }
    lds_fence();
    {
      const v8h lo = *(const v8h*)(&s_act[wv][nsub * MD + kb]);
      const v8h hi = *(const v8h*)(&s_act[wv][nsub * MD + 16 + kb]);
#pragma unroll
      for (int i = 0; i < 8; ++i) { a[i] = lo[i]; a[8 + i] = hi[i]; }
    }

    // ---- layer 2 -> message m
#pragma unroll
    for (int r = 0; r < 8; ++r) { c0[r] = bm2_0; c1[r] = bm2_1; }
    c0 = wmma_f16(a, B_m2_0, c0);
    c1 = wmma_f16(a, B_m2_1, c1);
    float m0[8], m1[8];
#pragma unroll
    for (int r = 0; r < 8; ++r) {
      m0[r] = tanh_fast(c0[r]);
      m1[r] = tanh_fast(c1[r]);
      s_act[wv][(mofs + r) * MD + n0] = (_Float16)m0[r];
      s_act[wv][(mofs + r) * MD + n1] = (_Float16)m1[r];
    }
    lds_fence();
    {
      const v8h lo = *(const v8h*)(&s_act[wv][nsub * MD + kb]);
      const v8h hi = *(const v8h*)(&s_act[wv][nsub * MD + 16 + kb]);
#pragma unroll
      for (int i = 0; i < 8; ++i) { a[i] = lo[i]; a[8 + i] = hi[i]; }
    }

    // ---- coord-gate layer: tanh(m@Wc1+bc1), then dot with Wc2 + tanh
#pragma unroll
    for (int r = 0; r < 8; ++r) { c0[r] = bc1_0; c1[r] = bc1_1; }
    c0 = wmma_f16(a, B_c1_0, c0);
    c1 = wmma_f16(a, B_c1_1, c1);

#pragma unroll
    for (int r = 0; r < 8; ++r) {
      float v = tanh_fast(c0[r]) * wc2_0 + tanh_fast(c1[r]) * wc2_1;
      v += __shfl_xor(v, 1, 32);          // butterfly within 16-lane half
      v += __shfl_xor(v, 2, 32);
      v += __shfl_xor(v, 4, 32);
      v += __shfl_xor(v, 8, 32);
      if (nsub == 0) s_cw[wv][mofs + r] = tanh_fast(v);
    }
    lds_fence();

    // ---- coordinate-update atomics (one lane per edge)
    if (!hiH && nsub < nv) {
      const float cw = s_cw[wv][nsub];
      atomicAdd(&num[ri * 3 + 0], rx * rinv * cw);
      atomicAdd(&num[ri * 3 + 1], ry * rinv * cw);
      atomicAdd(&num[ri * 3 + 2], rz * rinv * cw);
      if (addCnt) atomicAdd(&cnt[ri], 1.0f);   // degree is step-invariant
    }
    // ---- message aggregation: one atomic per (edge, feature), from D-layout
#pragma unroll
    for (int r = 0; r < 8; ++r) {
      const int mr = mofs + r;
      if (mr < nv) {
        const int er = s_row[wv][mr];
        atomicAdd(&agg[er * MD + n0], m0[r]);
        atomicAdd(&agg[er * MD + n1], m1[r]);
      }
    }
  }
}

// ---------------------------------------------------------------------------
// node kernel: residual MLP + NodeNorm + fire-rate masking (tiny vs edges)
// ---------------------------------------------------------------------------
__global__ __launch_bounds__(256) void egnca_node_kernel(
    const float* __restrict__ coord_in, const float* __restrict__ h_in,
    const float* __restrict__ num, const float* __restrict__ cnt,
    const float* __restrict__ agg,
    const float* __restrict__ Wn1, const float* __restrict__ bn1,
    const float* __restrict__ Wn2, const float* __restrict__ bn2,
    const float* __restrict__ mask_t,
    float* __restrict__ coord_out, float* __restrict__ h_out)
{
  __shared__ float sW1[(ND + MD) * MD];
  __shared__ float sb1[MD];
  __shared__ float sW2[MD * ND];
  __shared__ float sb2[ND];
  for (int i = threadIdx.x; i < (ND + MD) * MD; i += blockDim.x) sW1[i] = Wn1[i];
  for (int i = threadIdx.x; i < MD; i += blockDim.x) sb1[i] = bn1[i];
  for (int i = threadIdx.x; i < MD * ND; i += blockDim.x) sW2[i] = Wn2[i];
  for (int i = threadIdx.x; i < ND; i += blockDim.x) sb2[i] = bn2[i];
  __syncthreads();

  const int n = blockIdx.x * blockDim.x + threadIdx.x;
  if (n >= N_NODES) return;

  float hin[ND], ain[MD];
#pragma unroll
  for (int i = 0; i < ND; ++i) hin[i] = h_in[n * ND + i];
#pragma unroll
  for (int i = 0; i < MD; ++i) ain[i] = agg[n * MD + i];

  float t[MD];
#pragma unroll 4
  for (int j = 0; j < MD; ++j) {
    float acc = sb1[j];
#pragma unroll
    for (int i = 0; i < ND; ++i) acc += hin[i] * sW1[i * MD + j];
#pragma unroll
    for (int i = 0; i < MD; ++i) acc += ain[i] * sW1[(ND + i) * MD + j];
    t[j] = tanh_fast(acc);
  }

  float ho[ND];
  float mean = 0.0f;
#pragma unroll
  for (int i = 0; i < ND; ++i) {
    float acc = sb2[i];
#pragma unroll
    for (int j = 0; j < MD; ++j) acc += t[j] * sW2[j * ND + i];
    ho[i] = hin[i] + acc;
    mean += ho[i];
  }
  mean *= (1.0f / ND);
  float var = 0.0f;
#pragma unroll
  for (int i = 0; i < ND; ++i) {
    const float d = ho[i] - mean;
    var += d * d;
  }
  var *= (1.0f / ND);
  const float stdv  = __builtin_amdgcn_sqrtf(var + NODE_EPS);
  const float scale = __builtin_amdgcn_rsqf(stdv);   // x / std^(1/2)

  const float mk  = mask_t[n];
  const float omk = 1.0f - mk;
#pragma unroll
  for (int i = 0; i < ND; ++i)
    h_out[n * ND + i] = (ho[i] * scale) * mk + hin[i] * omk;

  const float rdeg = __builtin_amdgcn_rcpf(fmaxf(cnt[n], 1.0f));
#pragma unroll
  for (int k = 0; k < 3; ++k) {
    const float c0 = coord_in[n * 3 + k];
    const float cn = c0 + num[n * 3 + k] * rdeg;
    coord_out[n * 3 + k] = cn * mk + c0 * omk;
  }
}

// ---------------------------------------------------------------------------
// launcher: 2 EGNCA steps (zero -> edge -> node), ping-pong through d_ws
// ---------------------------------------------------------------------------
extern "C" void kernel_launch(void* const* d_in, const int* in_sizes, int n_in,
                              void* d_out, int out_size, void* d_ws, size_t ws_size,
                              hipStream_t stream) {
  const int*   eidx  = (const int*)d_in[0];
  const int*   row   = eidx;
  const int*   col   = eidx + N_EDGES;
  const float* crd0  = (const float*)d_in[1];
  const float* h0    = (const float*)d_in[2];
  const float* fire  = (const float*)d_in[3];
  const float* Wm1   = (const float*)d_in[4];
  const float* bm1   = (const float*)d_in[5];
  const float* Wm2   = (const float*)d_in[6];
  const float* bm2   = (const float*)d_in[7];
  const float* Wc1   = (const float*)d_in[8];
  const float* bc1   = (const float*)d_in[9];
  const float* Wc2   = (const float*)d_in[10];
  const float* Wn1   = (const float*)d_in[11];
  const float* bn1   = (const float*)d_in[12];
  const float* Wn2   = (const float*)d_in[13];
  const float* bn2   = (const float*)d_in[14];

  // ws layout: coordA | hA | cnt | num | agg   (num+agg contiguous for re-zero)
  float* ws     = (float*)d_ws;
  float* coordA = ws;                       // N*3
  float* hA     = coordA + N_NODES * 3;     // N*ND
  float* cnt    = hA + N_NODES * ND;        // N
  float* num    = cnt + N_NODES;            // N*3
  float* agg    = num + N_NODES * 3;        // N*MD

  float* coord_out = (float*)d_out;                 // N*3
  float* h_out     = (float*)d_out + N_NODES * 3;   // N*ND

  const int accAll  = N_NODES * (1 + 3 + MD);  // cnt|num|agg
  const int accStep = N_NODES * (3 + MD);      // num|agg only
  const int zgridA  = (accAll + 255) / 256;
  const int zgridS  = (accStep + 255) / 256;
  const int egrid   = 1024;                 // 8192 waves, grid-stride over 100K tiles
  const int ngrid   = (N_NODES + 255) / 256;

  // ---- step 0 ----
  egnca_zero_kernel<<<zgridA, 256, 0, stream>>>(cnt, accAll);
  egnca_edge_kernel<<<egrid, 256, 0, stream>>>(row, col, crd0, h0,
      Wm1, bm1, Wm2, bm2, Wc1, bc1, Wc2, num, cnt, agg, 1);
  egnca_node_kernel<<<ngrid, 256, 0, stream>>>(crd0, h0, num, cnt, agg,
      Wn1, bn1, Wn2, bn2, fire + 0 * N_NODES, coordA, hA);

  // ---- step 1 (cnt/degree reused from step 0) ----
  egnca_zero_kernel<<<zgridS, 256, 0, stream>>>(num, accStep);
  egnca_edge_kernel<<<egrid, 256, 0, stream>>>(row, col, coordA, hA,
      Wm1, bm1, Wm2, bm2, Wc1, bc1, Wc2, num, cnt, agg, 0);
  egnca_node_kernel<<<ngrid, 256, 0, stream>>>(coordA, hA, num, cnt, agg,
      Wn1, bn1, Wn2, bn2, fire + 1 * N_NODES, coord_out, h_out);
}
